// PagedAttentionGenerationModel_60790967108079
// MI455X (gfx1250) — matrix operand order repaired and checked by reading, not compiled, not run on hardware
//
#include <hip/hip_runtime.h>
#include <hip/hip_bf16.h>
#include <math.h>

typedef __attribute__((ext_vector_type(16))) _Float16 v16h;
typedef __attribute__((ext_vector_type(8)))  _Float16 h8;
typedef __attribute__((ext_vector_type(4)))  _Float16 h4;
typedef __attribute__((ext_vector_type(8)))  float    v8f;
typedef __attribute__((ext_vector_type(4)))  float    fvec4;

namespace {
constexpr int Bc = 4;        // batch
constexpr int Sc = 2048;     // seq len
constexpr int Hc = 8;        // kv heads
constexpr int Dc = 128;      // head dim
constexpr int Tc = Bc * Sc;  // total tokens
constexpr int QT = 16;       // q rows per wave tile
// base-2 softmax scale: (1/sqrt(128)) * log2(e)
constexpr float kScale2 = 0.08838834764831845f * 1.4426950408889634f;
}

// Broadcast: value of lane r -> lanes 0-15, lane r+8 -> lanes 16-31.
// Compile-time r => two v_readlane_b32 + one v_cndmask (no DS pipe).
__device__ __forceinline__ float bcast_rowpair(float x, int r, int hi)
{
  const float a = __int_as_float(__builtin_amdgcn_readlane(__float_as_int(x), r));
  const float b = __int_as_float(__builtin_amdgcn_readlane(__float_as_int(x), r + 8));
  return hi ? b : a;
}

// ---------------------------------------------------------------------------
// Pass 1a: K fp32 [t][h][d]  ->  f16 head-major kh[b][h][s][d]
// ---------------------------------------------------------------------------
__global__ __launch_bounds__(256) void cvt_k_f16(const float* __restrict__ ks,
                                                 _Float16* __restrict__ kh)
{
  const size_t i4 = ((size_t)blockIdx.x * 256 + threadIdx.x) * 4;
  const int d  = (int)(i4 & 127);
  const int s  = (int)((i4 >> 7) & (Sc - 1));
  const int hh = (int)((i4 >> 18) & (Hc - 1));
  const int bb = (int)(i4 >> 21);
  fvec4 f = *(const fvec4*)(ks + ((size_t)(bb * Sc + s) * Hc + hh) * Dc + d);
  h4 o;
  #pragma unroll
  for (int e = 0; e < 4; ++e) o[e] = (_Float16)f[e];
  *(h4*)(kh + i4) = o;
}

// ---------------------------------------------------------------------------
// Pass 1b: V fp32 [t][h][d] -> f16 transposed vt[b][h][d][s]
// 32x32 LDS tile transpose, coalesced global reads and writes.
// ---------------------------------------------------------------------------
__global__ __launch_bounds__(256) void cvt_v_t_f16(const float* __restrict__ vsrc,
                                                   _Float16* __restrict__ vt)
{
  const int tile = blockIdx.x;
  const int dt = tile & 3;            // d tile (4 x 32)
  const int st = (tile >> 2) & 63;    // s tile (64 x 32)
  const int h  = (tile >> 8) & (Hc - 1);
  const int b  = tile >> 11;
  const int s0 = st * 32, d0 = dt * 32;

  __shared__ _Float16 lt[32][36];     // +4 pad: no bank conflicts

  const int tid = threadIdx.x;
  {
    const int row = tid >> 3;             // s within tile
    const int dg  = (tid & 7) * 4;        // d within tile
    fvec4 f = *(const fvec4*)(vsrc + ((size_t)(b * Sc + s0 + row) * Hc + h) * Dc
                              + d0 + dg);
    #pragma unroll
    for (int e = 0; e < 4; ++e) lt[row][dg + e] = (_Float16)f[e];
  }
  __syncthreads();
  {
    const int drow = tid >> 3;            // d within tile
    const int sg   = (tid & 7) * 4;       // s within tile
    h4 o;
    #pragma unroll
    for (int e = 0; e < 4; ++e) o[e] = lt[sg + e][drow];
    *(h4*)(vt + ((size_t)((b * Hc + h) * Dc) + d0 + drow) * Sc + s0 + sg) = o;
  }
}

// ---------------------------------------------------------------------------
// One 32-key flash-attention tile. MASK is compile-time: the diagonal tile is
// peeled by the caller so the steady-state loop carries no mask code at all.
// ---------------------------------------------------------------------------
template <bool MASK>
__device__ __forceinline__ void fa_tile(
    int k0, int n, int hi, int qgl,
    const _Float16* __restrict__ kbase, const _Float16* __restrict__ vbase,
    const v16h* __restrict__ qb, v8f* __restrict__ acc,
    float& m_run, float& l_run)
{
  // ---- tile loads (K A-operand + V B-operand fragments, all b128)
  h8 kf[16];
  {
    const _Float16* kr0 = kbase + (size_t)(k0 + n) * Dc + 8 * hi;
    const _Float16* kr1 = kr0 + 16 * Dc;
    #pragma unroll
    for (int c = 0; c < 4; ++c) {
      kf[2 * c + 0] = *(const h8*)(kr0 + c * 32);
      kf[2 * c + 1] = *(const h8*)(kr0 + c * 32 + 16);
      kf[8 + 2 * c + 0] = *(const h8*)(kr1 + c * 32);
      kf[8 + 2 * c + 1] = *(const h8*)(kr1 + c * 32 + 16);
    }
  }
  h8 vf[16];
  #pragma unroll
  for (int dt = 0; dt < 8; ++dt) {
    const _Float16* vr = vbase + (size_t)(dt * 16 + n) * Sc + k0 + 8 * hi;
    vf[2 * dt + 0] = *(const h8*)(vr);
    vf[2 * dt + 1] = *(const h8*)(vr + 16);
  }

  // ---- S^T tiles: st0 = keys k0..k0+15, st1 = keys k0+16..k0+31
  v8f st0 = (v8f){0.f,0.f,0.f,0.f,0.f,0.f,0.f,0.f};
  v8f st1 = st0;
  #pragma unroll
  for (int c = 0; c < 4; ++c) {
    v16h ka0 = __builtin_shufflevector(kf[2 * c], kf[2 * c + 1],
                                       0,1,2,3,4,5,6,7,8,9,10,11,12,13,14,15);
    v16h ka1 = __builtin_shufflevector(kf[8 + 2 * c], kf[8 + 2 * c + 1],
                                       0,1,2,3,4,5,6,7,8,9,10,11,12,13,14,15);
    st0 = __builtin_amdgcn_wmma_f32_16x16x32_f16(false, ka0, false, qb[c],
                                                 (short)0, st0, false, false);
    st1 = __builtin_amdgcn_wmma_f32_16x16x32_f16(false, ka1, false, qb[c],
                                                 (short)0, st1, false, false);
  }

  // ---- optional causal mask (C-layout: reg r holds key k0 + r + 8*hi (+16))
  float s0a[8], s1a[8];
  #pragma unroll
  for (int r = 0; r < 8; ++r) {
    float a0 = st0[r];
    float a1 = st1[r];
    if (MASK) {
      const int key0 = k0 + r + 8 * hi;
      a0 = (key0      > qgl) ? -__builtin_inff() : a0;
      a1 = (key0 + 16 > qgl) ? -__builtin_inff() : a1;
    }
    s0a[r] = a0; s1a[r] = a1;
  }

  // ---- online softmax in base-2 (per-lane query; pair-reduce across hi)
  float tm = -__builtin_inff();
  #pragma unroll
  for (int r = 0; r < 8; ++r) tm = fmaxf(tm, fmaxf(s0a[r], s1a[r]));
  tm = fmaxf(tm, __shfl_xor(tm, 16, 32));
  const float m_new = fmaxf(m_run, tm);
  const float alpha = exp2f(m_run - m_new);

  v16h pf;                 // P directly in A-operand half layout
  float rs = 0.0f;
  #pragma unroll
  for (int r = 0; r < 8; ++r) {
    float e0 = exp2f(s0a[r] - m_new);
    float e1 = exp2f(s1a[r] - m_new);
    rs += e0 + e1;
    pf[r]     = (_Float16)e0;   // key = 8*hi + r
    pf[8 + r] = (_Float16)e1;   // key = 16 + 8*hi + r
  }
  rs += __shfl_xor(rs, 16, 32);
  l_run = l_run * alpha + rs;
  m_run = m_new;

  // ---- rescale accumulators only when some row's max actually moved
  // (alpha == exp2(0) == 1.0 exactly when unchanged)
#if __has_builtin(__builtin_amdgcn_ballot_w32)
  const bool need = __builtin_amdgcn_ballot_w32(alpha != 1.0f) != 0u;
#else
  const bool need = __ballot(alpha != 1.0f) != 0ull;
#endif
  if (need) {
    float ar[8];
    #pragma unroll
    for (int r = 0; r < 8; ++r) ar[r] = bcast_rowpair(alpha, r, hi);
    #pragma unroll
    for (int dt = 0; dt < 8; ++dt) {
      #pragma unroll
      for (int r = 0; r < 8; ++r) acc[dt][r] *= ar[r];
    }
  }

  // ---- O += P x V (V fragments already resident)
  #pragma unroll
  for (int dt = 0; dt < 8; ++dt) {
    v16h vb = __builtin_shufflevector(vf[2 * dt], vf[2 * dt + 1],
                                      0,1,2,3,4,5,6,7,8,9,10,11,12,13,14,15);
    acc[dt] = __builtin_amdgcn_wmma_f32_16x16x32_f16(false, pf, false, vb,
                                                     (short)0, acc[dt], false, false);
  }
}

// ---------------------------------------------------------------------------
// Pass 2: flash-attention prefill, one wave per (b, h, 16-row q-tile).
// S^T = K x Q^T so that P = exp2(S^T - m) lands directly in the 16-bit
// A-operand layout for O = P x V (no LDS, no cross-lane transpose).
// ---------------------------------------------------------------------------
__global__ __launch_bounds__(128) void pa_fa_fwd(
    const float* __restrict__ qp, const _Float16* __restrict__ kh,
    const _Float16* __restrict__ vt, float* __restrict__ op)
{
  const int lane = threadIdx.x & 31;
  const int wid  = threadIdx.x >> 5;
  const int task = blockIdx.x * 4 + wid;           // 0..4095
  const int qt = task & ((Sc / QT) - 1);           // q tile  (0..127)
  const int h  = (task >> 7) & (Hc - 1);           // head
  const int b  = task >> 10;                       // batch
  const int q0 = qt * QT;
  const int n  = lane & 15;
  const int hi = lane >> 4;

  // ---- Q as B-operand of S^T = K * Q^T (scores pre-scaled into base-2 units)
  const float* qrow = qp + ((size_t)(b * Sc + q0 + n) * Hc + h) * Dc;
  v16h qb[4];
  #pragma unroll
  for (int c = 0; c < 4; ++c) {
    const float* src = qrow + c * 32 + hi * 8;
    #pragma unroll
    for (int j = 0; j < 4; ++j) {
      fvec4 f = *(const fvec4*)(src + 4 * j + 4 * (j & 2));
      #pragma unroll
      for (int e = 0; e < 4; ++e) qb[c][4 * j + e] = (_Float16)(f[e] * kScale2);
    }
  }

  v8f acc[8];
  #pragma unroll
  for (int dt = 0; dt < 8; ++dt)
    acc[dt] = (v8f){0.f,0.f,0.f,0.f,0.f,0.f,0.f,0.f};

  float m_run = -__builtin_inff();
  float l_run = 0.0f;
  const int qgl = q0 + n;                  // this lane's query row
  const int nkt = (q0 + 47) >> 5;          // causal #key-tiles of 32

  const _Float16* kbase = kh + (size_t)(b * Hc + h) * Sc * Dc;
  const _Float16* vbase = vt + (size_t)(b * Hc + h) * Dc * Sc;

  // steady-state: all tiles strictly below the diagonal (no mask code)
  for (int kt = 0; kt < nkt - 1; ++kt) {
    const int k0 = kt << 5;
    __builtin_prefetch(kbase + (size_t)(k0 + 32 + lane) * Dc, 0, 1);
    __builtin_prefetch(vbase + (size_t)(lane * 4 + (hi ? 64 : 0)) * Sc + k0 + 32, 0, 1);
    fa_tile<false>(k0, n, hi, qgl, kbase, vbase, qb, acc, m_run, l_run);
  }
  // peeled diagonal tile (the only one needing the causal mask)
  fa_tile<true>((nkt - 1) << 5, n, hi, qgl, kbase, vbase, qb, acc, m_run, l_run);

  // ---- normalize and store: out[(b*S+q)*H*D + h*D + d]
  float lr[8];
  #pragma unroll
  for (int r = 0; r < 8; ++r) lr[r] = bcast_rowpair(l_run, r, hi);
  #pragma unroll
  for (int dt = 0; dt < 8; ++dt) {
    #pragma unroll
    for (int r = 0; r < 8; ++r) {
      const int qg = q0 + r + 8 * hi;
      op[((size_t)(b * Sc + qg) * Hc + h) * Dc + dt * 16 + n] =
          acc[dt][r] * __frcp_rn(lr[r]);
    }
  }
}

// positions / slot_mapping / block_tables are closed-form; harness output is
// float-typed, so emit the integer values as floats.
__global__ void pa_aux(float* __restrict__ aux)
{
  const int i = blockIdx.x * blockDim.x + threadIdx.x;
  if (i < Tc) {
    aux[i] = (float)(i & (Sc - 1));                       // positions
  } else if (i < 2 * Tc) {
    const int t = i - Tc;
    const int bb = t >> 11;                               // t / S
    const int r  = t & (Sc - 1);
    const int j  = r >> 7;                                // block within seq
    const int o  = r & 127;                               // offset in block
    const int blk = 64 - (bb + 1) * 16 + j;               // free-list pop order
    aux[i] = (float)(blk * 128 + o);                      // slot_mapping
  } else if (i < 2 * Tc + 64) {
    const int t  = i - 2 * Tc;
    const int bb = t >> 4;
    const int j  = t & 15;
    aux[i] = (float)(64 - (bb + 1) * 16 + j);             // block_tables
  }
}

extern "C" void kernel_launch(void* const* d_in, const int* in_sizes, int n_in,
                              void* d_out, int out_size, void* d_ws, size_t ws_size,
                              hipStream_t stream)
{
  (void)in_sizes; (void)n_in; (void)ws_size; (void)out_size;
  // inputs: [0] input_ids (int, unused), [1] q, [2] k, [3] v  (all fp32)
  const float* q = (const float*)d_in[1];
  const float* k = (const float*)d_in[2];
  const float* v = (const float*)d_in[3];
  float* out = (float*)d_out;

  // workspace: kh (f16, T*H*D) then vt (f16, T*H*D) = 33.5 MB total
  _Float16* kh = (_Float16*)d_ws;
  _Float16* vt = kh + (size_t)Tc * Hc * Dc;

  // Pass 1: one-time fp32 -> f16 conversion / transpose (O(S) work)
  const int nElem4 = (Tc * Hc * Dc) / 4;                  // 4 elems per thread
  cvt_k_f16<<<nElem4 / 256, 256, 0, stream>>>(k, kh);
  cvt_v_t_f16<<<Bc * Hc * 64 * 4, 256, 0, stream>>>(v, vt);

  // Pass 2: attention, 4096 wave tasks, 4 waves (128 threads) per block
  pa_fa_fwd<<<1024, 128, 0, stream>>>(q, kh, vt, out);

  // aux outputs appended after T*H*D floats
  float* aux = out + (size_t)Tc * Hc * Dc;
  const int auxN = 2 * Tc + 64;
  pa_aux<<<(auxN + 255) / 256, 256, 0, stream>>>(aux);
}
